// StyleLossPass2_20246475833348
// MI455X (gfx1250) — compile-verified, matrix operand order checked by reading, not run on hardware
//
#include <hip/hip_runtime.h>

typedef unsigned int u32;
typedef unsigned short u16;
typedef unsigned long long u64;

typedef __attribute__((ext_vector_type(16))) __bf16 v16bf;
typedef __attribute__((ext_vector_type(8)))  float  v8f;
typedef __attribute__((ext_vector_type(4)))  unsigned int u32x4;
typedef __attribute__((ext_vector_type(8)))  int i32x8;
typedef __attribute__((ext_vector_type(4)))  int i32x4;

#define C_CH 512
#define HW   4096        // 64*64
#define KP   4608        // 9 * 512 patch-K
#define PATCH_KSTEPS 144 // 4608 / 32
#define GRAM_KSTEPS  128 // 4096 / 32
#define LDSTRIDE 20      // dwords per LDS tile row (32 bf16 = 16 dwords + 4 pad)

#define ASYNC_LDS 1
#if defined(__has_builtin)
#if __has_builtin(__builtin_amdgcn_tensor_load_to_lds) && __has_builtin(__builtin_amdgcn_s_wait_tensorcnt)
#define USE_TDM 1
#endif
#endif
#ifndef USE_TDM
#define USE_TDM 0
#endif

union FragB { u32 u[8]; v16bf v; };

__device__ __forceinline__ u16 f2bf(float f) {
  union { float f; u32 u; } cv; cv.f = f;
  u32 u = cv.u;
  u32 r = u + 0x7FFFu + ((u >> 16) & 1u);       // round to nearest even
  if ((u & 0x7F800000u) == 0x7F800000u) r = u;  // inf/nan passthrough
  return (u16)(r >> 16);
}

__device__ __forceinline__ v8f zero_v8f() {
  v8f z;
#pragma unroll
  for (int i = 0; i < 8; ++i) z[i] = 0.0f;
  return z;
}

#if ASYNC_LDS
// 16B global -> LDS via gfx1250 async path (ASYNCcnt).
__device__ __forceinline__ void async_b128(u32 lds_addr, const void* gaddr) {
  asm volatile("global_load_async_to_lds_b128 %0, %1, off"
               :: "v"(lds_addr), "v"(gaddr) : "memory");
}
__device__ __forceinline__ void wait_async0() {
  asm volatile("s_wait_asynccnt 0x0" ::: "memory");
}
#endif

#if USE_TDM
// Tensor Data Mover: one descriptor moves a 128-row x 32 bf16 tile into LDS,
// padding each 64B row with 16B to produce the 20-dword LDS row stride.
// D# packing per CDNA5 ISA ss8.3/8.4 (2-D tensor,
// groups 2/3 zero).  6-arg builtin form (amdgpu-toolchain / clang-23).
__device__ __forceinline__ void tdm_tile_load(const u16* g, u32 lds_addr,
                                              u32 dim0, u32 dim1, u32 stride) {
  const u64 ga = (u64)(size_t)g;
  u32x4 g0;
  g0[0] = 1u;                                             // count=1, user D#
  g0[1] = lds_addr;                                       // lds_addr (bytes)
  g0[2] = (u32)ga;                                        // global_addr[31:0]
  g0[3] = ((u32)(ga >> 32) & 0x01FFFFFFu) | (2u << 30);   // ga[56:32] | type=2
  i32x8 g1;
  g1[0] = (int)((1u << 16)      // data_size = 2 bytes
              | (1u << 20)      // pad_enable
              | (3u << 22)      // pad_interval: 16 dwords (64B row)
              | (3u << 25));    // pad_amount: 4 dwords (16B)
  g1[1] = (int)((dim0 & 0xFFFFu) << 16);                  // tensor_dim0[15:0]
  g1[2] = (int)((dim0 >> 16) | ((dim1 & 0xFFFFu) << 16)); // dim0[31:16]|dim1[15:0]
  g1[3] = (int)((dim1 >> 16) | (32u << 16));              // dim1[31:16]|tile_dim0=32
  g1[4] = 128;                                            // tile_dim1=128, tile_dim2=0
  g1[5] = (int)stride;                                    // dim0_stride[31:0]
  g1[6] = 0;                                              // stride[47:32]|dim1_stride lo
  g1[7] = 0;
  const i32x4 gz4 = {0, 0, 0, 0};
  const i32x8 gz8 = {0, 0, 0, 0, 0, 0, 0, 0};
  __builtin_amdgcn_tensor_load_to_lds(g0, g1, gz4, gz4, gz8, 0);
}
__device__ __forceinline__ void wait_tensor0() {
  __builtin_amdgcn_s_wait_tensorcnt((short)0);
}
#endif

// ISA 7.12.2 16-bit A/B fragment layout: lane<16 -> K pairs {0..7,16..23},
// lane>=16 -> {8..15,24..31}; one K-pair (2 bf16) per dword.
__device__ __forceinline__ void load_frag(const u32* smem, int rowBase, int lane, FragB& f) {
  const int g = lane >> 4;
  const int m = rowBase + (lane & 15);
  const u32* p = smem + m * LDSTRIDE;
#pragma unroll
  for (int j = 0; j < 8; ++j) {
    const int d = (j < 4) ? (g * 4 + j) : (8 + g * 4 + (j - 4));
    f.u[j] = p[d];
  }
}

// ---------------------------------------------------------------- patches ---
__global__ __launch_bounds__(256) void build_patches_kernel(
    const float* __restrict__ fm, u16* __restrict__ P) {
  const size_t idx = (size_t)blockIdx.x * 256 + threadIdx.x;
  if (idx >= (size_t)HW * KP) return;
  const int i  = (int)(idx / KP);
  const int r  = (int)(idx % KP);
  const int dd = r >> 9;          // 0..8
  const int c  = r & 511;
  const int h  = i >> 6, w = i & 63;
  const int sh = h + (dd / 3) - 1;
  const int sw = w + (dd % 3) - 1;
  float v = 0.0f;
  if (sh >= 0 && sh < 64 && sw >= 0 && sw < 64)
    v = fm[((size_t)c << 12) + (sh << 6) + sw];
  P[idx] = f2bf(v);
}

// --------------------------------------------------------------- mask sum ---
__global__ __launch_bounds__(256) void mask_sum_kernel(
    const float* __restrict__ mask, float* __restrict__ msum) {
  __shared__ float red[256];
  float s = 0.0f;
  for (int i = threadIdx.x; i < HW; i += 256) s += mask[i];
  red[threadIdx.x] = s;
  __syncthreads();
  for (int off = 128; off > 0; off >>= 1) {
    if ((int)threadIdx.x < off) red[threadIdx.x] += red[threadIdx.x + off];
    __syncthreads();
  }
  if (threadIdx.x == 0) msum[0] = red[0] * (float)C_CH;
}

// -------------------------------------------- fused patch GEMM + argmax -----
__global__ __launch_bounds__(256) void match_argmax_kernel(
    const u16* __restrict__ Pc, const u16* __restrict__ Ps, int* __restrict__ corr) {
  __shared__ __align__(16) u32 Atile[2][128 * LDSTRIDE];
  __shared__ __align__(16) u32 Btile[2][128 * LDSTRIDE];
  __shared__ u64 keys[128];

  const int t    = threadIdx.x;
  const int lane = t & 31;
  const int wv   = t >> 5;
  const int wm   = wv & 3;   // 4 wave-rows * 32 rows
  const int wn   = wv >> 2;  // 2 wave-cols * 64 cols
  const int i0   = blockIdx.x * 128;
#if !USE_TDM
  const int row  = t >> 1;
  const int hf   = t & 1;
#endif

  if (t < 128) keys[t] = 0ull;
  __syncthreads();

  for (int j0 = 0; j0 < HW; j0 += 128) {
    v8f acc[2][4];
#pragma unroll
    for (int tm = 0; tm < 2; ++tm)
#pragma unroll
      for (int tn = 0; tn < 4; ++tn) acc[tm][tn] = zero_v8f();

#if USE_TDM
    if (wv == 0) {
      tdm_tile_load(Pc + (size_t)i0 * KP, (u32)(size_t)(&Atile[0][0]), KP, HW, KP);
      tdm_tile_load(Ps + (size_t)j0 * KP, (u32)(size_t)(&Btile[0][0]), KP, HW, KP);
    }
    for (int kt = 0; kt < PATCH_KSTEPS; ++kt) {
      if (wv == 0) wait_tensor0();
      __syncthreads();  // tile kt resident for all waves
      if (kt + 1 < PATCH_KSTEPS && wv == 0) {
        const int nb = (kt + 1) & 1;
        const int k0 = (kt + 1) * 32;
        tdm_tile_load(Pc + (size_t)i0 * KP + k0, (u32)(size_t)(&Atile[nb][0]), KP, HW, KP);
        tdm_tile_load(Ps + (size_t)j0 * KP + k0, (u32)(size_t)(&Btile[nb][0]), KP, HW, KP);
      }
      const u32* Ab = Atile[kt & 1];
      const u32* Bb = Btile[kt & 1];
#elif ASYNC_LDS
    __syncthreads();
    {
      const u32 la = (u32)(size_t)(&Atile[0][row * LDSTRIDE + hf * 8]);
      const u32 lb = (u32)(size_t)(&Btile[0][row * LDSTRIDE + hf * 8]);
      const u16* ga = Pc + (size_t)(i0 + row) * KP + hf * 16;
      const u16* gb = Ps + (size_t)(j0 + row) * KP + hf * 16;
      async_b128(la, ga);      async_b128(la + 16, ga + 8);
      async_b128(lb, gb);      async_b128(lb + 16, gb + 8);
    }
    for (int kt = 0; kt < PATCH_KSTEPS; ++kt) {
      wait_async0();
      __syncthreads();
      if (kt + 1 < PATCH_KSTEPS) {
        const int nb = (kt + 1) & 1;
        const int k0 = (kt + 1) * 32;
        const u32 la = (u32)(size_t)(&Atile[nb][row * LDSTRIDE + hf * 8]);
        const u32 lb = (u32)(size_t)(&Btile[nb][row * LDSTRIDE + hf * 8]);
        const u16* ga = Pc + (size_t)(i0 + row) * KP + k0 + hf * 16;
        const u16* gb = Ps + (size_t)(j0 + row) * KP + k0 + hf * 16;
        async_b128(la, ga);    async_b128(la + 16, ga + 8);
        async_b128(lb, gb);    async_b128(lb + 16, gb + 8);
      }
      const u32* Ab = Atile[kt & 1];
      const u32* Bb = Btile[kt & 1];
#else
    for (int kt = 0; kt < PATCH_KSTEPS; ++kt) {
      const int k0 = kt * 32;
      const uint4* ga = (const uint4*)(Pc + (size_t)(i0 + row) * KP + k0 + hf * 16);
      const uint4* gb = (const uint4*)(Ps + (size_t)(j0 + row) * KP + k0 + hf * 16);
      uint4 a0 = ga[0], a1 = ga[1];
      uint4 b0 = gb[0], b1 = gb[1];
      __syncthreads();
      uint4* sa = (uint4*)(&Atile[0][row * LDSTRIDE + hf * 8]);
      sa[0] = a0; sa[1] = a1;
      uint4* sb = (uint4*)(&Btile[0][row * LDSTRIDE + hf * 8]);
      sb[0] = b0; sb[1] = b1;
      __syncthreads();
      const u32* Ab = Atile[0];
      const u32* Bb = Btile[0];
#endif
      FragB af[2], bfr[4];
#pragma unroll
      for (int tm = 0; tm < 2; ++tm) load_frag(Ab, wm * 32 + tm * 16, lane, af[tm]);
#pragma unroll
      for (int tn = 0; tn < 4; ++tn) load_frag(Bb, wn * 64 + tn * 16, lane, bfr[tn]);
#pragma unroll
      for (int tm = 0; tm < 2; ++tm)
#pragma unroll
        for (int tn = 0; tn < 4; ++tn)
          acc[tm][tn] = __builtin_amdgcn_wmma_f32_16x16x32_bf16(
              false, af[tm].v, false, bfr[tn].v, (short)0, acc[tm][tn], false, false);
    }

    // Fused argmax epilogue. C/D layout: lane<16 -> (M=v, N=lane),
    // lane>=16 -> (M=v+8, N=lane-16). Ties pick smallest j (jnp.argmax).
#pragma unroll
    for (int tm = 0; tm < 2; ++tm) {
#pragma unroll
      for (int v = 0; v < 8; ++v) {
        float best = -3.402823466e38f;
        int   bj   = 0x7FFFFFFF;
#pragma unroll
        for (int tn = 0; tn < 4; ++tn) {
          float val = acc[tm][tn][v];
          int   j   = j0 + wn * 64 + tn * 16 + (lane & 15);
#pragma unroll
          for (int mk = 1; mk < 16; mk <<= 1) {
            const float ov = __shfl_xor(val, mk, 32);
            const int   oj = __shfl_xor(j, mk, 32);
            const bool take = (ov > val) | ((ov == val) & (oj < j));
            val = take ? ov : val;
            j   = take ? oj : j;
          }
          const bool take = (val > best) | ((val == best) & (j < bj));
          best = take ? val : best;
          bj   = take ? j   : bj;
        }
        if ((lane & 15) == 0) {
          union { float f; u32 u; } cv; cv.f = best;
          const u32 ord = (cv.u & 0x80000000u) ? ~cv.u : (cv.u | 0x80000000u);
          const u64 key = ((u64)ord << 32) | (u64)(0xFFFFFFFFu - (u32)bj);
          const int m = wm * 32 + tm * 16 + v + (lane >> 4) * 8;
          atomicMax(&keys[m], key);
        }
      }
    }
  }
  __syncthreads();
  if (t < 128) {
    const u64 k = keys[t];
    corr[i0 + t] = (int)(0xFFFFFFFFu - (u32)(k & 0xFFFFFFFFull));
  }
}

// ------------------------------------------------- gather + mask multiply ---
__global__ __launch_bounds__(256) void gather_mask_kernel(
    const float* __restrict__ input_fm, const float* __restrict__ style_fm,
    const int* __restrict__ corr, const float* __restrict__ mask,
    u16* __restrict__ Fi, u16* __restrict__ Fm) {
  const int idx = blockIdx.x * 256 + threadIdx.x;  // c*4096 + i
  const int i = idx & (HW - 1);
  const int c = idx >> 12;
  const float mk = mask[i];
  Fi[idx] = f2bf(input_fm[idx] * mk);
  Fm[idx] = f2bf(style_fm[((size_t)c << 12) + corr[i]] * mk);
}

// ------------------------------------------------------ gram: G = F @ F^T ---
__global__ __launch_bounds__(256) void gram_kernel(
    const u16* __restrict__ F, float* __restrict__ G) {
  __shared__ __align__(16) u32 Atile[2][128 * LDSTRIDE];
  __shared__ __align__(16) u32 Btile[2][128 * LDSTRIDE];

  const int t    = threadIdx.x;
  const int lane = t & 31;
  const int wv   = t >> 5;
  const int wm   = wv & 3;
  const int wn   = wv >> 2;
  const int bm   = (blockIdx.x >> 2) * 128;
  const int bn   = (blockIdx.x & 3) * 128;
  const int row  = t >> 1;
  const int hf   = t & 1;

  v8f acc[2][4];
#pragma unroll
  for (int tm = 0; tm < 2; ++tm)
#pragma unroll
    for (int tn = 0; tn < 4; ++tn) acc[tm][tn] = zero_v8f();

#if ASYNC_LDS
  {
    const u32 la = (u32)(size_t)(&Atile[0][row * LDSTRIDE + hf * 8]);
    const u32 lb = (u32)(size_t)(&Btile[0][row * LDSTRIDE + hf * 8]);
    const u16* ga = F + (size_t)(bm + row) * HW + hf * 16;
    const u16* gb = F + (size_t)(bn + row) * HW + hf * 16;
    async_b128(la, ga);      async_b128(la + 16, ga + 8);
    async_b128(lb, gb);      async_b128(lb + 16, gb + 8);
  }
  for (int kt = 0; kt < GRAM_KSTEPS; ++kt) {
    wait_async0();
    __syncthreads();
    if (kt + 1 < GRAM_KSTEPS) {
      const int nb = (kt + 1) & 1;
      const int k0 = (kt + 1) * 32;
      const u32 la = (u32)(size_t)(&Atile[nb][row * LDSTRIDE + hf * 8]);
      const u32 lb = (u32)(size_t)(&Btile[nb][row * LDSTRIDE + hf * 8]);
      const u16* ga = F + (size_t)(bm + row) * HW + k0 + hf * 16;
      const u16* gb = F + (size_t)(bn + row) * HW + k0 + hf * 16;
      async_b128(la, ga);    async_b128(la + 16, ga + 8);
      async_b128(lb, gb);    async_b128(lb + 16, gb + 8);
    }
    const u32* Ab = Atile[kt & 1];
    const u32* Bb = Btile[kt & 1];
#else
  for (int kt = 0; kt < GRAM_KSTEPS; ++kt) {
    const int k0 = kt * 32;
    const uint4* ga = (const uint4*)(F + (size_t)(bm + row) * HW + k0 + hf * 16);
    const uint4* gb = (const uint4*)(F + (size_t)(bn + row) * HW + k0 + hf * 16);
    uint4 a0 = ga[0], a1 = ga[1];
    uint4 b0 = gb[0], b1 = gb[1];
    __syncthreads();
    uint4* sa = (uint4*)(&Atile[0][row * LDSTRIDE + hf * 8]);
    sa[0] = a0; sa[1] = a1;
    uint4* sb = (uint4*)(&Btile[0][row * LDSTRIDE + hf * 8]);
    sb[0] = b0; sb[1] = b1;
    __syncthreads();
    const u32* Ab = Atile[0];
    const u32* Bb = Btile[0];
#endif
    FragB af[2], bfr[4];
#pragma unroll
    for (int tm = 0; tm < 2; ++tm) load_frag(Ab, wm * 32 + tm * 16, lane, af[tm]);
#pragma unroll
    for (int tn = 0; tn < 4; ++tn) load_frag(Bb, wn * 64 + tn * 16, lane, bfr[tn]);
#pragma unroll
    for (int tm = 0; tm < 2; ++tm)
#pragma unroll
      for (int tn = 0; tn < 4; ++tn)
        acc[tm][tn] = __builtin_amdgcn_wmma_f32_16x16x32_bf16(
            false, af[tm].v, false, bfr[tn].v, (short)0, acc[tm][tn], false, false);
  }

#pragma unroll
  for (int tm = 0; tm < 2; ++tm)
#pragma unroll
    for (int tn = 0; tn < 4; ++tn)
#pragma unroll
      for (int v = 0; v < 8; ++v) {
        const int r = bm + wm * 32 + tm * 16 + v + (lane >> 4) * 8;
        const int c = bn + wn * 64 + tn * 16 + (lane & 15);
        G[(size_t)r * C_CH + c] = acc[tm][tn][v];
      }
}

// ----------------------------------------------------------- final loss -----
__global__ __launch_bounds__(256) void final_loss_kernel(
    const float* __restrict__ G1, const float* __restrict__ G2,
    const float* __restrict__ msum, float* __restrict__ out) {
  __shared__ float red[256];
  const float inv = 1.0f / msum[0];
  float s = 0.0f;
  for (int i = threadIdx.x; i < C_CH * C_CH; i += 256) {
    const float d = (G1[i] - G2[i]) * inv;
    s += d * d;
  }
  red[threadIdx.x] = s;
  __syncthreads();
  for (int off = 128; off > 0; off >>= 1) {
    if ((int)threadIdx.x < off) red[threadIdx.x] += red[threadIdx.x + off];
    __syncthreads();
  }
  if (threadIdx.x == 0)
    out[0] = red[0] * (100.0f / (float)(C_CH * C_CH));
}

// ---------------------------------------------------------------- launch ----
extern "C" void kernel_launch(void* const* d_in, const int* in_sizes, int n_in,
                              void* d_out, int out_size, void* d_ws, size_t ws_size,
                              hipStream_t stream) {
  (void)in_sizes; (void)n_in; (void)out_size; (void)ws_size;
  const float* input_fm   = (const float*)d_in[0];
  const float* style_fm   = (const float*)d_in[1];
  const float* content_fm = (const float*)d_in[2];
  const float* mask       = (const float*)d_in[3];
  float* out = (float*)d_out;

  char* ws = (char*)d_ws;
  size_t off = 0;
  u16*   Pc   = (u16*)(ws + off);   off += (size_t)HW * KP * 2;      // 37.75 MB
  u16*   Ps   = (u16*)(ws + off);   off += (size_t)HW * KP * 2;      // 37.75 MB
  u16*   Fi   = (u16*)(ws + off);   off += (size_t)C_CH * HW * 2;    // 4 MB
  u16*   Fm   = (u16*)(ws + off);   off += (size_t)C_CH * HW * 2;    // 4 MB
  float* G1   = (float*)(ws + off); off += (size_t)C_CH * C_CH * 4;  // 1 MB
  float* G2   = (float*)(ws + off); off += (size_t)C_CH * C_CH * 4;  // 1 MB
  int*   corr = (int*)(ws + off);   off += (size_t)HW * 4;           // 16 KB
  float* msum = (float*)(ws + off); off += 256;

  const int pb = (int)(((size_t)HW * KP + 255) / 256);
  build_patches_kernel<<<pb, 256, 0, stream>>>(content_fm, Pc);
  build_patches_kernel<<<pb, 256, 0, stream>>>(style_fm, Ps);
  mask_sum_kernel<<<1, 256, 0, stream>>>(mask, msum);
  match_argmax_kernel<<<HW / 128, 256, 0, stream>>>(Pc, Ps, corr);
  gather_mask_kernel<<<(C_CH * HW) / 256, 256, 0, stream>>>(
      input_fm, style_fm, corr, mask, Fi, Fm);
  gram_kernel<<<16, 256, 0, stream>>>(Fi, G1);
  gram_kernel<<<16, 256, 0, stream>>>(Fm, G2);
  final_loss_kernel<<<1, 256, 0, stream>>>(G1, G2, msum, out);
}
